// NonparametricAttentionPooling_41188736369116
// MI455X (gfx1250) — matrix operand order repaired and checked by reading, not compiled, not run on hardware
//
#include <hip/hip_runtime.h>
#include <hip/hip_bf16.h>
#include <math.h>

// Problem constants (reference: B=8, C=256, N=2048)
#define B_ 8
#define C_ 256
#define N_ 2048
#define BN_EPS 1e-5f

typedef __attribute__((ext_vector_type(16))) __bf16 v16bf;
typedef __attribute__((ext_vector_type(8)))  __bf16 v8bf;
typedef __attribute__((ext_vector_type(8)))  float  v8f;
typedef __attribute__((ext_vector_type(4)))  unsigned int u32x4;
typedef __attribute__((ext_vector_type(8)))  int i32x8;
typedef __attribute__((ext_vector_type(4)))  int i32x4;

// ---------------------------------------------------------------------------
// WMMA wrapper: D = A(16x32 bf16) * B(32x16 bf16) + C(16x16 f32)
// ---------------------------------------------------------------------------
__device__ __forceinline__ v8f wmma_bf16(v16bf a, v16bf b, v8f c) {
  return __builtin_amdgcn_wmma_f32_16x16x32_bf16(
      /*neg_a=*/false, a, /*neg_b=*/false, b,
      /*c_mod=*/(short)0, c, /*reuse_a=*/false, /*reuse_b=*/false);
}

// A-matrix (16x32, bf16) fragment per ISA layout (row = lane&15):
//   lanes 0-15 : VGPR0-3 K=0..7,  VGPR4-7 K=16..23
//   lanes 16-31: VGPR0-3 K=8..15, VGPR4-7 K=24..31
__device__ __forceinline__ v16bf load_fragA(const __bf16* rowbase, int half) {
  const v8bf lo = *(const v8bf*)(rowbase + 8 * half);
  const v8bf hi = *(const v8bf*)(rowbase + 16 + 8 * half);
  v16bf r;
#pragma unroll
  for (int i = 0; i < 8; ++i) { r[i] = lo[i]; r[i + 8] = hi[i]; }
  return r;
}

// B-matrix (32x16, bf16) fragment, lane's column contiguous in K
// (source row-major [col][k]): lanes 0-15 K=0..7,16..23, lanes 16-31 K=8..15,24..31.
__device__ __forceinline__ v16bf load_fragB_contig(const __bf16* rowbase, int half) {
  const v8bf lo = *(const v8bf*)(rowbase + 8 * half);
  const v8bf hi = *(const v8bf*)(rowbase + 16 + 8 * half);
  v16bf r;
#pragma unroll
  for (int i = 0; i < 8; ++i) { r[i] = lo[i]; r[i + 8] = hi[i]; }
  return r;
}

// ---------------------------------------------------------------------------
// Tensor Data Mover: 2D tile load (global -> LDS), D# built per ISA 8.3/8.4.
// 6-arg builtin form (clang-23 / therock-10.0).
// ---------------------------------------------------------------------------
__device__ __forceinline__ void tdm_load_2d(
    unsigned lds_off, const void* gptr,
    unsigned tensor_d0, unsigned tensor_d1,
    unsigned tile_d0, unsigned tile_d1,
    unsigned long long stride0) {
  unsigned long long ga = (unsigned long long)gptr;
  u32x4 g0;
  g0[0] = 1u;                                   // count=1, user descriptor
  g0[1] = lds_off;                              // lds_addr [63:32]
  g0[2] = (unsigned)(ga & 0xFFFFFFFFu);         // global_addr lo
  g0[3] = (unsigned)((ga >> 32) & 0x1FFFFFFu)   // global_addr[56:32]
          | (2u << 30);                         // type=2 (image)
  i32x8 g1;
  g1[0] = (int)(1u << 16);                      // data_size=1 -> 2 bytes
  g1[1] = (int)((tensor_d0 & 0xFFFFu) << 16);   // tensor_dim0[15:0] @ [63:48]
  g1[2] = (int)((tensor_d0 >> 16) | ((tensor_d1 & 0xFFFFu) << 16));
  g1[3] = (int)((tensor_d1 >> 16) | (tile_d0 << 16));   // tile_dim0 @ [127:112]
  g1[4] = (int)(tile_d1 & 0xFFFFu);             // tile_dim1; tile_dim2=0
  g1[5] = (int)(stride0 & 0xFFFFFFFFull);       // tensor_dim0_stride lo
  g1[6] = (int)((stride0 >> 32) & 0xFFFFull);   // stride0 hi; dim1_stride=0
  g1[7] = 0;
  i32x4 z4 = {0, 0, 0, 0};
  i32x8 z8 = {0, 0, 0, 0, 0, 0, 0, 0};
  __builtin_amdgcn_tensor_load_to_lds(g0, g1, z4, z4, z8, 0);
}

__device__ __forceinline__ unsigned lds_offset_of(const void* p) {
  // Generic pointers into LDS carry the LDS byte offset in the low 32 bits
  // (ISA 10.2).
  return (unsigned)(unsigned long long)p;
}

// ---------------------------------------------------------------------------
// Kernel 0: bf16 copies of x — ptsT (B,N,C) and xbf (B,C,N) — plus per-point
// squared norms; zero the per-channel BN accumulators (every launch).
// ---------------------------------------------------------------------------
__global__ void nap_prep_kernel(const float* __restrict__ x,
                                __bf16* __restrict__ ptsT,
                                __bf16* __restrict__ xbf,
                                float* __restrict__ sq,
                                float* __restrict__ chanSum,
                                float* __restrict__ chanSumSq) {
  const int n = blockIdx.x;
  const int b = blockIdx.y;
  const int c = threadIdx.x;  // 0..255
  if (b == 0 && n == 0) { chanSum[c] = 0.0f; chanSumSq[c] = 0.0f; }
  const size_t xi = ((size_t)b * C_ + c) * N_ + n;
  const float v = x[xi];
  ptsT[((size_t)b * N_ + n) * C_ + c] = (__bf16)v;
  xbf[xi] = (__bf16)v;
  __shared__ float red[C_];
  red[c] = v * v;
  __syncthreads();
  for (int s = C_ / 2; s > 0; s >>= 1) {
    if (c < s) red[c] += red[c + s];
    __syncthreads();
  }
  if (c == 0) sq[b * N_ + n] = red[0];
}

// ---------------------------------------------------------------------------
// Kernel 1: fused Gram -> Gaussian-mix weights -> unnormalized W*pts + row
// sums, one pass over m.  8 waves/WG: wave pair shares one 16-row n-tile
// (Gram redundant), each wave owns 8 c-tiles.  Tile staging via TDM.
// ---------------------------------------------------------------------------
__global__ __launch_bounds__(256) void nap_pool_kernel(
    const __bf16* __restrict__ ptsT, const __bf16* __restrict__ xbf,
    const float* __restrict__ sq, float* __restrict__ nf,
    float* __restrict__ chanSum, float* __restrict__ chanSumSq) {

  __shared__ __bf16 Xm[32 * C_];       // 16 KB: pts rows m0..m0+31 (row-major in c)
  __shared__ __bf16 Xc[C_ * 32];       // 16 KB: same tile, x-layout (row-major in m)
  __shared__ __bf16 Wlds[4][16 * 32];  //  4 KB: per-pair weight tile staging

  const int tid  = threadIdx.x;
  const int wave = tid >> 5;
  const int lane = tid & 31;
  const int half = lane >> 4;
  const int l16  = lane & 15;
  const int pair = wave >> 1;          // 0..3 : n-tile within block
  const int sub  = wave & 1;           // 0/1  : c-tile half

  const int b  = blockIdx.y;
  const int n0 = (blockIdx.x * 4 + pair) * 16;

  const __bf16* ptsBt = ptsT + (size_t)b * N_ * C_;
  const __bf16* xbfBt = xbf + (size_t)b * C_ * N_;
  const float*  sqBt  = sq + b * N_;

  // Register-resident A fragments for this wave's n-tile (16 x 256 bf16).
  v16bf a_reg[8];
  {
    const __bf16* arow = ptsBt + (size_t)(n0 + l16) * C_;
#pragma unroll
    for (int k = 0; k < 8; ++k) a_reg[k] = load_fragA(arow + 32 * k, half);
  }

  float sqA[8];
#pragma unroll
  for (int v = 0; v < 8; ++v) sqA[v] = sqBt[n0 + v + 8 * half];

  v8f acc[8];
#pragma unroll
  for (int t = 0; t < 8; ++t)
    acc[t] = (v8f){0.f, 0.f, 0.f, 0.f, 0.f, 0.f, 0.f, 0.f};
  float rowAcc[8] = {0.f, 0.f, 0.f, 0.f, 0.f, 0.f, 0.f, 0.f};

  const unsigned xm_off = lds_offset_of(&Xm[0]);
  const unsigned xc_off = lds_offset_of(&Xc[0]);

  for (int m0 = 0; m0 < N_; m0 += 32) {
    __syncthreads();  // previous iteration done with Xm/Xc/Wlds
    if (wave == 0) {
      // pts[m0:m0+32, 0:256]  ->  Xm[32][256]
      tdm_load_2d(xm_off, ptsBt + (size_t)m0 * C_,
                  /*tensor_d0=*/C_, /*tensor_d1=*/N_,
                  /*tile_d0=*/C_, /*tile_d1=*/32, /*stride0=*/C_);
      // x[b, 0:256, m0:m0+32] ->  Xc[256][32]
      tdm_load_2d(xc_off, xbfBt + m0,
                  /*tensor_d0=*/N_, /*tensor_d1=*/C_,
                  /*tile_d0=*/32, /*tile_d1=*/C_, /*stride0=*/N_);
      __builtin_amdgcn_s_wait_tensorcnt(0);
    }
    __syncthreads();  // staged tiles visible to all waves

    // ---- Gram tiles + Gaussian mixture weights (two 16x16 subtiles) ----
#pragma unroll
    for (int s = 0; s < 2; ++s) {
      v8f g = (v8f){0.f, 0.f, 0.f, 0.f, 0.f, 0.f, 0.f, 0.f};
      const __bf16* brow = &Xm[(s * 16 + l16) * C_];
#pragma unroll
      for (int k = 0; k < 8; ++k)
        g = wmma_bf16(a_reg[k], load_fragB_contig(brow + 32 * k, half), g);
      const float sqm = sqBt[m0 + s * 16 + l16];
#pragma unroll
      for (int v = 0; v < 8; ++v) {
        float d2 = fmaxf(sqA[v] + sqm - 2.0f * g[v], 0.0f);
        // bw=0.5: 0.5*(d/bw)^2 = 2*d2 ; bw*2 -> d2/2 ; bw/2 -> 8*d2
        float w = 0.5f * __expf(-2.0f * d2) +
                  0.3f * __expf(-0.5f * d2) +
                  0.2f * __expf(-8.0f * d2);
        rowAcc[v] += w;
        if (sub == 0)
          Wlds[pair][(v + 8 * half) * 32 + s * 16 + l16] = (__bf16)w;
      }
    }
    __syncthreads();  // Wlds visible to pair mate

    // ---- GEMM2: U += W(16x32) * pts(32x16) for this wave's 8 c-tiles ----
    v16bf wa = load_fragA(&Wlds[pair][l16 * 32], half);
#pragma unroll
    for (int tt = 0; tt < 8; ++tt) {
      const int t = 8 * sub + tt;
      v16bf bb = load_fragB_contig(&Xc[(16 * t + l16) * 32], half);
      acc[tt] = wmma_bf16(wa, bb, acc[tt]);
    }
  }

  // Row-sum reduction across each 16-lane group -> 1/(S + 1e-8).
#pragma unroll
  for (int v = 0; v < 8; ++v) {
    float r = rowAcc[v];
    r += __shfl_xor(r, 1);
    r += __shfl_xor(r, 2);
    r += __shfl_xor(r, 4);
    r += __shfl_xor(r, 8);
    rowAcc[v] = 1.0f / (r + 1e-8f);
  }

  // Normalize, write new_f (B,C,N), accumulate per-channel BN stats.
#pragma unroll
  for (int tt = 0; tt < 8; ++tt) {
    const int c = 16 * (8 * sub + tt) + l16;
    float ps = 0.f, pq = 0.f;
#pragma unroll
    for (int v = 0; v < 8; ++v) {
      const float val = acc[tt][v] * rowAcc[v];
      const int n = n0 + v + 8 * half;
      nf[((size_t)b * C_ + c) * N_ + n] = val;
      ps += val;
      pq += val * val;
    }
    atomicAdd(&chanSum[c], ps);
    atomicAdd(&chanSumSq[c], pq);
  }
}

// ---------------------------------------------------------------------------
// Kernel 2: BatchNorm (batch stats) + exact GELU, elementwise over (B,C,N).
// ---------------------------------------------------------------------------
__global__ void nap_bn_gelu_kernel(const float* __restrict__ nf,
                                   const float* __restrict__ chanSum,
                                   const float* __restrict__ chanSumSq,
                                   const float* __restrict__ gamma,
                                   const float* __restrict__ beta,
                                   float* __restrict__ out) {
  const size_t i = (size_t)blockIdx.x * blockDim.x + threadIdx.x;
  const int c = (int)((i / N_) % C_);
  const float inv_cnt = 1.0f / (float)(B_ * N_);
  const float mean = chanSum[c] * inv_cnt;
  const float var  = chanSumSq[c] * inv_cnt - mean * mean;
  float y = (nf[i] - mean) * rsqrtf(var + BN_EPS);
  y = y * gamma[c] + beta[c];
  out[i] = 0.5f * y * (1.0f + erff(y * 0.70710678118654752f));
}

// ---------------------------------------------------------------------------
extern "C" void kernel_launch(void* const* d_in, const int* in_sizes, int n_in,
                              void* d_out, int out_size, void* d_ws, size_t ws_size,
                              hipStream_t stream) {
  (void)in_sizes; (void)n_in; (void)out_size; (void)ws_size;
  const float* x     = (const float*)d_in[0];
  const float* gamma = (const float*)d_in[1];
  const float* beta  = (const float*)d_in[2];
  float* out = (float*)d_out;

  char* ws = (char*)d_ws;
  __bf16* ptsT = (__bf16*)ws;                         // 8 MB  (B,N,C) bf16
  __bf16* xbf  = ptsT + (size_t)B_ * N_ * C_;         // 8 MB  (B,C,N) bf16
  float* sq    = (float*)(xbf + (size_t)B_ * C_ * N_);// 64 KB
  float* chanSum   = sq + B_ * N_;                    // 1 KB
  float* chanSumSq = chanSum + C_;                    // 1 KB
  float* nf        = chanSumSq + C_;                  // 16 MB

  nap_prep_kernel<<<dim3(N_, B_), C_, 0, stream>>>(x, ptsT, xbf, sq, chanSum, chanSumSq);
  nap_pool_kernel<<<dim3(N_ / 64, B_), 256, 0, stream>>>(ptsT, xbf, sq, nf, chanSum, chanSumSq);
  nap_bn_gelu_kernel<<<(B_ * C_ * N_) / 256, 256, 0, stream>>>(
      nf, chanSum, chanSumSq, gamma, beta, out);
}
